// CrossAttention_65644280152073
// MI455X (gfx1250) — matrix operand order repaired and checked by reading, not compiled, not run on hardware
//
#include <hip/hip_runtime.h>
#include <hip/hip_bf16.h>

// CrossAttention with relative position embeddings, CDNA5 / gfx1250 (wave32, WMMA + TDM).
// B=5120, T=16, C=512, H=8, Dh=64. All GEMMs via v_wmma_f32_16x16x32_bf16;
// B-operand / v staging via TENSOR_LOAD_TO_LDS (Tensor Data Mover).

typedef __bf16 bf16;
typedef __attribute__((ext_vector_type(16))) __bf16 v16bf;
typedef __attribute__((ext_vector_type(8)))  __bf16 v8bf;
typedef __attribute__((ext_vector_type(4)))  __bf16 v4bf;
typedef __attribute__((ext_vector_type(8)))  float  v8f;
typedef __attribute__((ext_vector_type(4)))  float  v4f;
typedef unsigned int u32x4 __attribute__((ext_vector_type(4)));
typedef int          i32x8 __attribute__((ext_vector_type(8)));
typedef int          i32x4 __attribute__((ext_vector_type(4)));

#if __has_builtin(__builtin_amdgcn_tensor_load_to_lds) && __has_builtin(__builtin_amdgcn_s_wait_tensorcnt)
#define HAVE_TDM 1
#else
#define HAVE_TDM 0
#endif

static __device__ __forceinline__ v16bf cat8(v8bf a, v8bf b) {
    return __builtin_shufflevector(a, b, 0,1,2,3,4,5,6,7,8,9,10,11,12,13,14,15);
}
static __device__ __forceinline__ v8f wmma_bf16(v16bf a, v16bf b, v8f c) {
    // (neg_a, A, neg_b, B, c_mod, C, reuse_a, reuse_b)
    return __builtin_amdgcn_wmma_f32_16x16x32_bf16(false, a, false, b, (short)0, c, false, false);
}
static __device__ __forceinline__ v8f zero8() {
    v8f z = {0.f,0.f,0.f,0.f,0.f,0.f,0.f,0.f};
    return z;
}
// Low 32 bits of a generic pointer into LDS == LDS byte offset (ISA 10.2 aperture rule).
static __device__ __forceinline__ unsigned lds_off(const void* p) {
    return (unsigned)(unsigned long long)p;
}

#if HAVE_TDM
// 2D TDM load: tile_d0 x tile_d1 elements of 2 bytes, global row stride = stride0
// (elements), into LDS at lds_addr. Optional LDS padding: +4 DWORDs per 256 DWORDs.
static __device__ __forceinline__ void tdm_load_2d_bf16(unsigned lds_addr,
                                                        const void* gptr,
                                                        unsigned tile_d0, unsigned tile_d1,
                                                        unsigned stride0, bool pad) {
    unsigned long long ga = (unsigned long long)gptr;
    u32x4 g0;
    g0[0] = 1u;                                        // count=1, user mode
    g0[1] = lds_addr;                                  // lds_addr (bytes)
    g0[2] = (unsigned)(ga & 0xFFFFFFFFu);              // global_addr[31:0]
    g0[3] = (unsigned)((ga >> 32) & 0x01FFFFFFu) | 0x80000000u; // addr[56:32] | type=2
    i32x8 g1;
    unsigned w0 = (1u << 16);                          // data_size=1 (2 bytes)
    if (pad) w0 |= (1u << 20) | (7u << 22) | (3u << 25); // pad_en, every 256 DW, +4 DW
    g1[0] = (int)w0;
    g1[1] = (int)(tile_d0 << 16);                      // tensor_dim0[15:0] in [63:48]
    g1[2] = (int)(tile_d1 << 16);                      // dim0[31:16]=0 | tensor_dim1[15:0]
    g1[3] = (int)(tile_d0 << 16);                      // dim1[31:16]=0 | tile_dim0
    g1[4] = (int)(tile_d1 & 0xFFFFu);                  // tile_dim1 | tile_dim2=0
    g1[5] = (int)stride0;                              // tensor_dim0_stride[31:0]
    g1[6] = 0;                                         // stride0[47:32] | stride1[15:0]
    g1[7] = 0;
    i32x4 gz4 = {0, 0, 0, 0};
    i32x8 gz8 = {0, 0, 0, 0, 0, 0, 0, 0};
    // 6-arg form (this toolchain): (g0, g1, g2, g3, g4, cpol)
    __builtin_amdgcn_tensor_load_to_lds(g0, g1, gz4, gz4, gz8, 0);
}
#endif

// ---------------------------------------------------------------------------
// Conversions / layout prep
// ---------------------------------------------------------------------------
__global__ __launch_bounds__(256) void cvt_x(const float* __restrict__ x,
                                             bf16* __restrict__ xh, int n) {
    int tid = blockIdx.x * blockDim.x + threadIdx.x;
    int stride = gridDim.x * blockDim.x;
    for (int idx = tid; idx * 4 < n; idx += stride) {
        v4f xv = *(const v4f*)(x + (size_t)idx * 4);
        v4bf o;
        #pragma unroll
        for (int e = 0; e < 4; e++) o[e] = (bf16)xv[e];
        *(v4bf*)(xh + (size_t)idx * 4) = o;
    }
}

// Pre-transposed weights, Bt[n][k] = W[k][n]  (W stored (in,out) row-major)
__global__ __launch_bounds__(256) void build_wt(const float* __restrict__ Wq,
                                                const float* __restrict__ Wk,
                                                const float* __restrict__ Wv,
                                                const float* __restrict__ Wout,
                                                bf16* __restrict__ Btqkv,
                                                bf16* __restrict__ Btout) {
    int i = blockIdx.x * blockDim.x + threadIdx.x;   // 512*512
    if (i >= 512 * 512) return;
    int c = i >> 9;          // output col
    int k = i & 511;         // input dim (contiguous per thread -> coalesced stores)
    Btqkv[((size_t)(0 * 512 + c)) * 512 + k] = (bf16)Wq[(size_t)k * 512 + c];
    Btqkv[((size_t)(1 * 512 + c)) * 512 + k] = (bf16)Wk[(size_t)k * 512 + c];
    Btqkv[((size_t)(2 * 512 + c)) * 512 + k] = (bf16)Wv[(size_t)k * 512 + c];
    Btout[((size_t)c) * 512 + k]             = (bf16)Wout[(size_t)k * 512 + c];
}

// Pre-swizzle rel tables into WMMA B-fragment per-lane order.
// tksw[tt][ks][lane][e] : B(K=d, N=t) = relK[t][d]; frag(t-tile tt, k-step ks)
// tvsw[nt][lane][e]     : B(K=t, N=d) = relV[t][d]; single K-step (t = 0..31)
__global__ __launch_bounds__(256) void build_tbl(const float* __restrict__ relk,
                                                 const float* __restrict__ relv,
                                                 bf16* __restrict__ tksw,
                                                 bf16* __restrict__ tvsw) {
    for (int idx = threadIdx.x; idx < 2048; idx += 256) {
        int e  = idx & 15;
        int L  = (idx >> 4) & 31;
        int ks = (idx >> 9) & 1;
        int tt = (idx >> 10) & 1;
        int trow = tt * 16 + (L & 15);
        int d    = 32 * ks + ((L & 16) ? 16 : 0) + e;
        tksw[idx] = (bf16)relk[(size_t)trow * 64 + d];
    }
    for (int idx = threadIdx.x; idx < 2048; idx += 256) {
        int e  = idx & 15;
        int L  = (idx >> 4) & 31;
        int nt = (idx >> 9) & 3;
        int trow = e + ((L & 16) ? 16 : 0);
        int d    = nt * 16 + (L & 15);
        tvsw[idx] = (bf16)relv[(size_t)trow * 64 + d];
    }
}

// ---------------------------------------------------------------------------
// GEMM: C(MxN) = A(MxK bf16 row-major) @ Bt(NxK bf16 row-major)^T
// Block: 128x128; 8 waves, each wave 16(M) x 128(N); K-step 32.
// Bt slab (128 x 512) staged to LDS via TDM (padded rows: 520 elems ->
// 16-lane ds_load_b128 fragment reads are bank-conflict free).
// OUTF32=0: store bf16. OUTF32=1: store f32 + bias.
// ---------------------------------------------------------------------------
#define BROW 520   // padded LDS row stride in elements (512 + 8)

template<int OUTF32>
__global__ __launch_bounds__(256) void gemm_bf16(const bf16* __restrict__ A,
                                                 const bf16* __restrict__ Bt,
                                                 void* __restrict__ Cv,
                                                 const float* __restrict__ bias,
                                                 int M, int N, int K, int nbn) {
    extern __shared__ __align__(16) char smem_raw[];
    bf16* bsh = (bf16*)smem_raw;                     // [128][BROW]

    const int lane = threadIdx.x & 31;
    const int wave = threadIdx.x >> 5;
    const int nb   = blockIdx.x % nbn;
    const int mb   = blockIdx.x / nbn;
    const int m0   = mb * 128 + wave * 16;
    const int n0   = nb * 128;
    const int l15  = lane & 15;
    const int hi   = (lane >> 4) & 1;

    // ---- Stage Bt[n0 .. n0+127][0..K) into LDS ----
    const bf16* bslab = Bt + (size_t)n0 * K;
#if HAVE_TDM
    if (wave == 0) {
        // 2D tile: K elems per row, 128 rows, contiguous (stride = K), padded in LDS.
        tdm_load_2d_bf16(lds_off(bsh), (const void*)bslab,
                         (unsigned)K, 128u, (unsigned)K, /*pad=*/true);
        __builtin_amdgcn_s_wait_tensorcnt(0);
    }
#else
    {
        int chunks_per_row = K / 8;                  // 8 bf16 per 16B chunk
        for (int idx = (int)threadIdx.x; idx < 128 * chunks_per_row; idx += 256) {
            int row = idx / chunks_per_row;
            int c   = (idx % chunks_per_row) * 8;
            *(v8bf*)(bsh + (size_t)row * BROW + c) =
                *(const v8bf*)(bslab + (size_t)row * K + c);
        }
    }
#endif
    __syncthreads();

    v8f acc[8];
    #pragma unroll
    for (int i = 0; i < 8; i++) acc[i] = zero8();

    // A fragment: lane row = l15 (both 16-lane halves cover M=0..15),
    // K chunk: [hi*16 .. +15] contiguous (matches B fragment)
    const bf16* arow = A + (size_t)(m0 + l15) * K + hi * 16;

    #pragma unroll 2
    for (int k0 = 0; k0 < K; k0 += 32) {
        __builtin_prefetch((const void*)(arow + k0 + 128), 0, 0); // next A tiles
        v8bf a0 = *(const v8bf*)(arow + k0);
        v8bf a1 = *(const v8bf*)(arow + k0 + 8);
        v16bf afrag = cat8(a0, a1);
        #pragma unroll
        for (int nt = 0; nt < 8; nt++) {
            // B fragment from LDS: lane col = l15, K = k0 + hi*16 + e (contig 16)
            const bf16* brow = bsh + (size_t)(nt * 16 + l15) * BROW + k0 + hi * 16;
            v16bf bfrag = cat8(*(const v8bf*)brow, *(const v8bf*)(brow + 8));
            acc[nt] = wmma_bf16(afrag, bfrag, acc[nt]);
        }
    }

    // D layout: lane col n = n0+nt*16+l15; VGPR r -> row m0 + r + hi*8
    #pragma unroll
    for (int nt = 0; nt < 8; nt++) {
        int n = n0 + nt * 16 + l15;
        #pragma unroll
        for (int r = 0; r < 8; r++) {
            int m = m0 + r + hi * 8;
            if constexpr (OUTF32) {
                ((float*)Cv)[(size_t)m * N + n] = acc[nt][r] + bias[n];
            } else {
                ((bf16*)Cv)[(size_t)m * N + n] = (bf16)acc[nt][r];
            }
        }
    }
}

// ---------------------------------------------------------------------------
// Attention core: block = one batch b, wave = head h. qkv rows: [b*16+n][1536].
//  sim = (q@k^T + gather(p)) * 1/8 ; p = q @ relK^T (16x32)
//  out = attn@v + W@relV, W[i][t] = attn[i][i+t-16]
// v for all 8 heads (16x512) staged by one TDM load per block.
// ---------------------------------------------------------------------------
__global__ __launch_bounds__(256) void attn_kernel(const bf16* __restrict__ qkv,
                                                   const bf16* __restrict__ tksw,
                                                   const bf16* __restrict__ tvsw,
                                                   bf16* __restrict__ ctx) {
    __shared__ __align__(16) float pshm[8][16][32];   // 16 KB (per-wave regions)
    __shared__ __align__(16) bf16  ashm[8][16][16];   //  4 KB (per-wave regions)
    __shared__ __align__(16) bf16  vshm[16][512];     // 16 KB (block-shared: all heads)

    const int lane = threadIdx.x & 31;
    const int wave = threadIdx.x >> 5;
    const int b    = blockIdx.x;
    const int h    = wave;
    const int l15  = lane & 15;
    const int hi   = (lane >> 4) & 1;

    const bf16* rowb = qkv + (size_t)b * 16 * 1536;
    const bf16* qb = rowb + h * 64;
    const bf16* kb = rowb + 512 + h * 64;
    const bf16* vball = rowb + 1024;                  // v block, all heads

    // ---- Stage v (16 x 512, row stride 1536) into LDS ----
#if HAVE_TDM
    if (wave == 0) {
        tdm_load_2d_bf16(lds_off(&vshm[0][0]), (const void*)vball,
                         512u, 16u, 1536u, /*pad=*/false);
        __builtin_amdgcn_s_wait_tensorcnt(0);
    }
#else
    for (int idx = (int)threadIdx.x; idx < 16 * 64; idx += 256) {
        int j = idx >> 6;
        int c = (idx & 63) * 8;
        *(v8bf*)&vshm[j][c] = *(const v8bf*)(vball + (size_t)j * 1536 + c);
    }
#endif
    __syncthreads();

    // Q fragments (A operand, M=i, K=d), two K-steps of 32.
    const bf16* qr = qb + (size_t)l15 * 1536 + hi * 16;
    v16bf q0 = cat8(*(const v8bf*)(qr),      *(const v8bf*)(qr + 8));
    v16bf q1 = cat8(*(const v8bf*)(qr + 32), *(const v8bf*)(qr + 40));

    // K^T fragments (B operand: K=d, N=j). Contiguous 16 bf16 per lane.
    const bf16* kr = kb + (size_t)l15 * 1536 + hi * 16;
    v16bf kt0 = cat8(*(const v8bf*)(kr),      *(const v8bf*)(kr + 8));
    v16bf kt1 = cat8(*(const v8bf*)(kr + 32), *(const v8bf*)(kr + 40));

    v8f sim = zero8();
    sim = wmma_bf16(q0, kt0, sim);
    sim = wmma_bf16(q1, kt1, sim);

    // p[i][t] = q_i . relK[t]  via pre-swizzled B frags; spill to LDS for gather.
    #pragma unroll
    for (int tt = 0; tt < 2; tt++) {
        const bf16* t0 = tksw + ((size_t)(tt * 2 + 0) * 32 + lane) * 16;
        const bf16* t1 = tksw + ((size_t)(tt * 2 + 1) * 32 + lane) * 16;
        v16bf b0 = cat8(*(const v8bf*)t0, *(const v8bf*)(t0 + 8));
        v16bf b1 = cat8(*(const v8bf*)t1, *(const v8bf*)(t1 + 8));
        v8f p = zero8();
        p = wmma_bf16(q0, b0, p);
        p = wmma_bf16(q1, b1, p);
        #pragma unroll
        for (int r = 0; r < 8; r++)
            pshm[wave][r + hi * 8][tt * 16 + l15] = p[r];
    }

    // sim[i][j] += p[i][j-i+16]; scale; softmax over j (16 lanes = one row).
    float av[8];
    #pragma unroll
    for (int r = 0; r < 8; r++) {
        int i = r + hi * 8;
        int t = l15 - i + 16;              // in [1,31]
        av[r] = (sim[r] + pshm[wave][i][t]) * 0.125f;
    }
    #pragma unroll
    for (int r = 0; r < 8; r++) {
        float m = av[r];
        m = fmaxf(m, __shfl_xor(m, 8, 16));
        m = fmaxf(m, __shfl_xor(m, 4, 16));
        m = fmaxf(m, __shfl_xor(m, 2, 16));
        m = fmaxf(m, __shfl_xor(m, 1, 16));
        float e = __expf(av[r] - m);
        float s = e;
        s += __shfl_xor(s, 8, 16);
        s += __shfl_xor(s, 4, 16);
        s += __shfl_xor(s, 2, 16);
        s += __shfl_xor(s, 1, 16);
        av[r] = e / s;
        ashm[wave][r + hi * 8][l15] = (bf16)av[r];
    }

    // Build A fragments from LDS: attn (K=j<16, rest 0) and W (K=t 0..31).
    v16bf aAV, aW;
    #pragma unroll
    for (int e = 0; e < 16; e++) {
        int kk = e + hi * 16;                          // contiguous A K-index
        int i  = l15;
        aAV[e] = (kk < 16) ? ashm[wave][i][kk] : (bf16)0.f;
        int jj = i + kk - 16;
        aW[e]  = (jj >= 0 && jj < 16) ? ashm[wave][i][jj] : (bf16)0.f;
    }

    // out tiles over d (4 x 16): attn@v + W@relV, then store ctx[b*16+i][h*64+d].
    #pragma unroll
    for (int nt = 0; nt < 4; nt++) {
        int d = nt * 16 + l15;
        v16bf vB;
        #pragma unroll
        for (int e = 0; e < 16; e++) {
            int kk = e + hi * 16;                      // contiguous B K-index (matches aAV)
            vB[e] = (kk < 16) ? vshm[kk][h * 64 + d] : (bf16)0.f;
        }
        const bf16* tv = tvsw + ((size_t)nt * 32 + lane) * 16;
        v16bf tvB = cat8(*(const v8bf*)tv, *(const v8bf*)(tv + 8));
        v8f o = zero8();
        o = wmma_bf16(aAV, vB, o);
        o = wmma_bf16(aW, tvB, o);
        #pragma unroll
        for (int r = 0; r < 8; r++) {
            int i = r + hi * 8;
            ctx[((size_t)(b * 16 + i)) * 512 + h * 64 + d] = (bf16)o[r];
        }
    }
}

// ---------------------------------------------------------------------------
extern "C" void kernel_launch(void* const* d_in, const int* in_sizes, int n_in,
                              void* d_out, int out_size, void* d_ws, size_t ws_size,
                              hipStream_t stream) {
    const float* x    = (const float*)d_in[0];
    const float* Wq   = (const float*)d_in[1];
    const float* Wk   = (const float*)d_in[2];
    const float* Wv   = (const float*)d_in[3];
    const float* Wout = (const float*)d_in[4];
    const float* bout = (const float*)d_in[5];
    const float* relk = (const float*)d_in[6];
    const float* relv = (const float*)d_in[7];
    float* out = (float*)d_out;

    const int BT = 5120 * 16;  // 81920 rows

    char* ws = (char*)d_ws;
    size_t off = 0;
    auto carve = [&](size_t bytes) -> char* {
        char* p = ws + off;
        off += (bytes + 255) & ~(size_t)255;
        return p;
    };
    bf16* xh    = (bf16*)carve((size_t)BT * 512 * 2);    //  80 MB
    bf16* qkv   = (bf16*)carve((size_t)BT * 1536 * 2);   // 240 MB
    bf16* ctx   = (bf16*)carve((size_t)BT * 512 * 2);    //  80 MB
    bf16* Btqkv = (bf16*)carve((size_t)1536 * 512 * 2);
    bf16* Btout = (bf16*)carve((size_t)512 * 512 * 2);
    bf16* tksw  = (bf16*)carve(2048 * 2);
    bf16* tvsw  = (bf16*)carve(2048 * 2);

    cvt_x<<<4096, 256, 0, stream>>>(x, xh, BT * 512);
    build_wt<<<1024, 256, 0, stream>>>(Wq, Wk, Wv, Wout, Btqkv, Btout);
    build_tbl<<<1, 256, 0, stream>>>(relk, relv, tksw, tvsw);

    const size_t gemm_lds = (size_t)128 * BROW * 2;      // 133,120 B (<320 KB/WGP)

    // QKV projection: 81920 x 1536 = xh @ Btqkv^T
    gemm_bf16<0><<<640 * 12, 256, gemm_lds, stream>>>(xh, Btqkv, (void*)qkv, nullptr,
                                                      BT, 1536, 512, 12);
    // Attention core: one block per batch, one wave per head
    attn_kernel<<<5120, 256, 0, stream>>>(qkv, tksw, tvsw, ctx);
    // Output projection + bias: 81920 x 512 f32
    gemm_bf16<1><<<640 * 4, 256, gemm_lds, stream>>>(ctx, Btout, (void*)out, bout,
                                                     BT, 512, 512, 4);
}
